// ProbSFNO_81046032876052
// MI455X (gfx1250) — matrix operand order repaired and checked by reading, not compiled, not run on hardware
//
#include <hip/hip_runtime.h>
#include <hip/hip_bf16.h>
#include <math.h>

typedef __attribute__((ext_vector_type(16))) _Float16 v16h;
typedef __attribute__((ext_vector_type(8)))  float    v8f;

#define NB    4
#define INC   5
#define OUTC  2
#define NLAT  121
#define NLON  240
#define LW    48
#define EMB   256
#define H2    512
#define SP    (NLAT*NLON)          // 29040
#define TWO_PI 6.2831853071795864769f

// LDS row strides (halves), 16B-aligned and bank-spread across 16 rows
#define AS_STRIDE 264              // for K=256 tiles
#define AC_STRIDE 520              // for K=512 tiles

// ---------------------------------------------------------------------------
// A-fragment from an LDS row (16-bit A 16x32 layout, ISA 7.12.2):
// lanes 0-15: K = kbase+{0..7} and kbase+{16..23}; lanes 16-31: +8.
// ---------------------------------------------------------------------------
__device__ __forceinline__ v16h a_frag(const _Float16* __restrict__ row,
                                       int kbase, bool hi) {
  v16h av;
  const int koff = hi ? 8 : 0;
  #pragma unroll
  for (int v = 0; v < 8; ++v) {
    int kk = (v < 4 ? 2*v : 16 + 2*(v - 4)) + koff;
    av[2*v]     = row[kbase + kk];
    av[2*v + 1] = row[kbase + kk + 1];
  }
  return av;
}

// ---------------------------------------------------------------------------
// B-fragment (32x16) gathered from global with row stride strideK floats;
// the N column is folded into `base`.  lanes 0-15: K=0..7,16..23; 16-31: +8.
// ---------------------------------------------------------------------------
__device__ __forceinline__ v16h b_frag(const float* __restrict__ base,
                                       int kbase, bool hi, int strideK) {
  v16h bv;
  const int koff = hi ? 8 : 0;
  #pragma unroll
  for (int v = 0; v < 8; ++v) {
    int k0 = kbase + koff + (v < 4 ? 2*v : 16 + 2*(v - 4));
    bv[2*v]     = (_Float16)base[(size_t)k0      * strideK];
    bv[2*v + 1] = (_Float16)base[(size_t)(k0 + 1)* strideK];
  }
  return bv;
}

// ---------------------------------------------------------------------------
// prep: lon-means of x, m=0 Legendre projection a0[b,c,l], S[l], Pbar[l],
//       Wmix = w_out @ w_in, zero dcum.  One block, 256 threads.
// ---------------------------------------------------------------------------
__global__ __launch_bounds__(256) void sfno_prep(
    const float* __restrict__ x, const float* __restrict__ P,
    const float* __restrict__ wq, const float* __restrict__ w_in,
    const float* __restrict__ b_in, const float* __restrict__ w_out,
    float* __restrict__ a0, float* __restrict__ Svec,
    float* __restrict__ Pbar, float* __restrict__ dcum,
    float* __restrict__ Wmix)
{
  __shared__ float xbar[NB*INC*NLAT];   // 2420
  __shared__ float q0[NB*INC*LW];       // 960
  __shared__ float Tl[LW];
  const int tid = threadIdx.x;

  // phase 1: xbar[b,c,j] = mean over lon of x
  for (int idx = tid; idx < NB*INC*NLAT; idx += 256) {
    int b = idx / (INC*NLAT);
    int r = idx % (INC*NLAT);
    int c = r / NLAT, j = r % NLAT;
    const float* px = x + ((size_t)(b*INC + c)*NLAT + j)*NLON;
    float s = 0.f;
    for (int w = 0; w < NLON; ++w) s += px[w];
    xbar[idx] = s * (1.0f/NLON);
  }
  __syncthreads();

  // phase 2a: q0[b,c,l] = sum_j xbar[b,c,j] * P[l,0,j] * wq[j]
  for (int idx = tid; idx < NB*INC*LW; idx += 256) {
    int b = idx / (INC*LW);
    int r = idx % (INC*LW);
    int c = r / LW, l = r % LW;
    const float* pp = P + (size_t)l*LW*NLAT;   // m = 0 slice
    const float* xb = &xbar[(b*INC + c)*NLAT];
    float s = 0.f;
    for (int j = 0; j < NLAT; ++j) s += xb[j]*pp[j]*wq[j];
    q0[idx] = s;
  }
  // phase 2b: T[l], S[l] = 2*pi*T[l], Pbar[l] = mean_j P[l,0,j]
  if (tid < LW) {
    const float* pp = P + (size_t)tid*LW*NLAT;
    float t = 0.f, pb = 0.f;
    for (int j = 0; j < NLAT; ++j) { t += pp[j]*wq[j]; pb += pp[j]; }
    Tl[tid]   = t;
    Svec[tid] = TWO_PI * t;
    Pbar[tid] = pb * (1.0f/NLAT);
  }
  __syncthreads();

  // phase 3: a0[b,c,l] = 2*pi*( sum_c' w_in[c,c']*q0[b,c',l] + b_in[c]*T[l] )
  for (int idx = tid; idx < NB*EMB*LW; idx += 256) {
    int b = idx / (EMB*LW);
    int r = idx % (EMB*LW);
    int c = r / LW, l = r % LW;
    float s = b_in[c]*Tl[l];
    for (int cp = 0; cp < INC; ++cp)
      s += w_in[c*INC + cp]*q0[(b*INC + cp)*LW + l];
    a0[idx] = TWO_PI * s;
  }

  // dcum = 0
  for (int idx = tid; idx < NB*EMB; idx += 256) dcum[idx] = 0.f;

  // Wmix[o,c'] = sum_c w_out[o,c]*w_in[c,c']
  if (tid < 2*OUTC*INC) {
    int o = tid / INC, cp = tid % INC;
    float s = 0.f;
    for (int c = 0; c < EMB; ++c) s += w_out[o*EMB + c]*w_in[c*INC + cp];
    Wmix[tid] = s;
  }
}

// ---------------------------------------------------------------------------
// filter_gemm: block l in [0,48): f[b,o,l] = sum_i a[b,i,l]*filt_r[i,o,l]
// a[b,i,l] = a0[b,i,l] + dcum[b,i]*S[l].  A staged in LDS (f16, zero-padded
// M 4->16), then WMMA f32<-f16.  256 threads = 8 waves, 2 o-tiles each.
// ---------------------------------------------------------------------------
__global__ __launch_bounds__(256) void sfno_filter_gemm(
    const float* __restrict__ a0, const float* __restrict__ dcum,
    const float* __restrict__ Svec, const float* __restrict__ filt_r,
    float* __restrict__ f, int layer)
{
  __shared__ _Float16 As[16*AS_STRIDE];
  const int l    = blockIdx.x;
  const int tid  = threadIdx.x;
  const int lane = tid & 31;
  const int wave = tid >> 5;
  const int m    = lane & 15;
  const bool hi  = lane >= 16;
  const float Sl = Svec[l];
  const float* fr = filt_r + (size_t)layer*EMB*EMB*LW;

  // stage A tile: As[m][i] = a[m,i,l] for m<4 else 0  (coalesced over i)
  for (int idx = tid; idx < 16*EMB; idx += 256) {
    int mm = idx >> 8, i = idx & (EMB-1);
    float v = 0.f;
    if (mm < NB)
      v = a0[((mm*EMB) + i)*LW + l] + dcum[mm*EMB + i]*Sl;
    As[mm*AS_STRIDE + i] = (_Float16)v;
  }
  __syncthreads();

  const _Float16* arow = &As[m*AS_STRIDE];
  for (int ot = 0; ot < 2; ++ot) {
    const int otile = wave*2 + ot;
    const int n = otile*16 + m;
    const float* bbase = fr + (size_t)n*LW + l;   // element (i,n,l); stride_i = EMB*LW
    v8f acc = {};
    for (int ks = 0; ks < 8; ++ks) {
      v16h av = a_frag(arow, ks*32, hi);
      v16h bv = b_frag(bbase, ks*32, hi, EMB*LW);
      acc = __builtin_amdgcn_wmma_f32_16x16x32_f16(
          false, av, false, bv, (short)0, acc, false, false);
    }
    // C/D layout: lanes 0-15 -> N=lane, VGPR r -> M=r; b occupies M=0..3
    if (!hi) {
      #pragma unroll
      for (int r = 0; r < NB; ++r)
        f[((size_t)(r*EMB) + otile*16 + m)*LW + l] = acc[r];
    }
  }
}

// ---------------------------------------------------------------------------
// mlp: g = f . Pbar ; delta = gelu(g@w1+b1)@w2+b2 ; dcum += delta.
// Both GEMMs on WMMA (M padded 4->16, A tiles staged f16 in LDS).
// Last layer also computes ov[b,o] = b_out[o] + w_out.(b_in + dcum).
// One block, 256 threads = 8 waves.
// ---------------------------------------------------------------------------
__global__ __launch_bounds__(256) void sfno_mlp(
    const float* __restrict__ f, const float* __restrict__ Pbar,
    const float* __restrict__ w1, const float* __restrict__ b1,
    const float* __restrict__ w2, const float* __restrict__ b2,
    const float* __restrict__ w_out, const float* __restrict__ b_out,
    const float* __restrict__ b_in,
    float* __restrict__ dcum, float* __restrict__ ov, int layer)
{
  __shared__ _Float16 Gs[16*AS_STRIDE];   // g, f16, 16x256 (rows 4..15 = 0)
  __shared__ _Float16 Act[16*AC_STRIDE];  // gelu(g@w1+b1), f16, 16x512
  __shared__ float    dsh[NB*EMB];        // updated dcum for ov
  const int tid  = threadIdx.x;
  const int lane = tid & 31;
  const int wave = tid >> 5;
  const int m    = lane & 15;
  const bool hi  = lane >= 16;

  for (int idx = tid; idx < 16*AS_STRIDE; idx += 256) Gs[idx]  = (_Float16)0.f;
  for (int idx = tid; idx < 16*AC_STRIDE; idx += 256) Act[idx] = (_Float16)0.f;
  __syncthreads();

  // g[b,o] = sum_l f[b,o,l]*Pbar[l]
  for (int idx = tid; idx < NB*EMB; idx += 256) {
    const float* pf = f + (size_t)idx*LW;
    float s = 0.f;
    for (int l = 0; l < LW; ++l) s += pf[l]*Pbar[l];
    Gs[(idx >> 8)*AS_STRIDE + (idx & (EMB-1))] = (_Float16)s;
  }
  __syncthreads();

  // GEMM1: t = g @ w1 + b1 ; Act = gelu(t)   (K=256, N=512)
  const float* W1 = w1 + (size_t)layer*EMB*H2;
  const float* B1 = b1 + layer*H2;
  {
    const _Float16* arow = &Gs[m*AS_STRIDE];
    for (int nt = wave*4; nt < wave*4 + 4; ++nt) {
      const int h = nt*16 + m;
      const float* bbase = W1 + h;          // element (o,h); stride_o = H2
      v8f acc = {};
      for (int ks = 0; ks < 8; ++ks) {
        v16h av = a_frag(arow, ks*32, hi);
        v16h bv = b_frag(bbase, ks*32, hi, H2);
        acc = __builtin_amdgcn_wmma_f32_16x16x32_f16(
            false, av, false, bv, (short)0, acc, false, false);
      }
      if (!hi) {
        float bb = B1[h];
        #pragma unroll
        for (int r = 0; r < NB; ++r) {
          float s = acc[r] + bb;
          Act[r*AC_STRIDE + h] =
              (_Float16)(0.5f*s*(1.0f + erff(s*0.70710678118654752f)));
        }
      }
    }
  }
  __syncthreads();

  // GEMM2: delta = Act @ w2 + b2 ; dcum += delta   (K=512, N=256)
  const float* W2 = w2 + (size_t)layer*H2*EMB;
  const float* B2 = b2 + layer*EMB;
  {
    const _Float16* arow = &Act[m*AC_STRIDE];
    for (int nt = wave*2; nt < wave*2 + 2; ++nt) {
      const int c = nt*16 + m;
      const float* bbase = W2 + c;          // element (h,c); stride_h = EMB
      v8f acc = {};
      for (int ks = 0; ks < 16; ++ks) {
        v16h av = a_frag(arow, ks*32, hi);
        v16h bv = b_frag(bbase, ks*32, hi, EMB);
        acc = __builtin_amdgcn_wmma_f32_16x16x32_f16(
            false, av, false, bv, (short)0, acc, false, false);
      }
      if (!hi) {
        float bb = B2[c];
        #pragma unroll
        for (int r = 0; r < NB; ++r) {
          int idx = r*EMB + c;
          float nd = dcum[idx] + acc[r] + bb;
          dcum[idx] = nd;
          dsh[idx]  = nd;
        }
      }
    }
  }

  if (layer == 3) {
    __syncthreads();
    if (tid < NB*2*OUTC) {
      int b = tid >> 2, o = tid & 3;
      float s = b_out[o];
      for (int c = 0; c < EMB; ++c)
        s += w_out[o*EMB + c]*(b_in[c] + dsh[b*EMB + c]);
      ov[tid] = s;
    }
  }
}

// ---------------------------------------------------------------------------
// final: out[b,o,p] = sum_c Wmix[o,c]*x[b,c,p] + ov[b,o];
// sample = mu + eps*exp(log_sigma).  d_out = [sample | mu | log_sigma].
// ---------------------------------------------------------------------------
__global__ __launch_bounds__(256) void sfno_final(
    const float* __restrict__ x, const float* __restrict__ eps,
    const float* __restrict__ Wmix, const float* __restrict__ ov,
    float* __restrict__ out)
{
  __shared__ float sW[2*OUTC*INC];
  __shared__ float sov[NB*2*OUTC];
  const int tid = threadIdx.x;
  if (tid < 2*OUTC*INC) sW[tid]  = Wmix[tid];
  if (tid < NB*2*OUTC)  sov[tid] = ov[tid];
  __syncthreads();

  int idx = blockIdx.x*blockDim.x + tid;
  if (idx >= NB*SP) return;
  int b = idx / SP, p = idx % SP;

  float xc[INC];
  #pragma unroll
  for (int c = 0; c < INC; ++c)
    xc[c] = x[((size_t)(b*INC + c))*SP + p];

  float v[2*OUTC];
  #pragma unroll
  for (int o = 0; o < 2*OUTC; ++o) {
    float s = sov[b*2*OUTC + o];
    #pragma unroll
    for (int c = 0; c < INC; ++c) s += sW[o*INC + c]*xc[c];
    v[o] = s;
  }

  const size_t CH = (size_t)NB*OUTC*SP;   // 232320 per output tensor
  #pragma unroll
  for (int o = 0; o < OUTC; ++o) {
    size_t off = ((size_t)(b*OUTC + o))*SP + p;
    float mu = v[o], ls = v[OUTC + o];
    float e  = eps[off];
    out[off]        = mu + e*expf(ls);   // sample
    out[CH + off]   = mu;                // mu
    out[2*CH + off] = ls;                // log_sigma
  }
}

// ---------------------------------------------------------------------------
extern "C" void kernel_launch(void* const* d_in, const int* in_sizes, int n_in,
                              void* d_out, int out_size, void* d_ws, size_t ws_size,
                              hipStream_t stream) {
  (void)in_sizes; (void)n_in; (void)out_size; (void)ws_size;
  const float* x      = (const float*)d_in[0];
  const float* eps    = (const float*)d_in[1];
  const float* Pmat   = (const float*)d_in[2];
  const float* wq     = (const float*)d_in[3];
  const float* w_in   = (const float*)d_in[4];
  const float* b_in   = (const float*)d_in[5];
  const float* filt_r = (const float*)d_in[6];
  // d_in[7] = filt_i : provably dead (only Re reaches the outputs)
  const float* w1     = (const float*)d_in[8];
  const float* b1     = (const float*)d_in[9];
  const float* w2     = (const float*)d_in[10];
  const float* b2     = (const float*)d_in[11];
  const float* w_out  = (const float*)d_in[12];
  const float* b_out  = (const float*)d_in[13];

  float* ws   = (float*)d_ws;
  float* a0   = ws;                 // 4*256*48 = 49152
  float* Svec = a0   + NB*EMB*LW;   // 48
  float* Pbar = Svec + LW;          // 48
  float* dcum = Pbar + LW;          // 1024
  float* Wmix = dcum + NB*EMB;      // 20
  float* ov   = Wmix + 2*OUTC*INC;  // 16
  float* f    = ov   + NB*2*OUTC;   // 49152

  sfno_prep<<<1, 256, 0, stream>>>(x, Pmat, wq, w_in, b_in, w_out,
                                   a0, Svec, Pbar, dcum, Wmix);
  for (int i = 0; i < 4; ++i) {
    sfno_filter_gemm<<<LW, 256, 0, stream>>>(a0, dcum, Svec, filt_r, f, i);
    sfno_mlp<<<1, 256, 0, stream>>>(f, Pbar, w1, b1, w2, b2,
                                    w_out, b_out, b_in, dcum, ov, i);
  }
  sfno_final<<<(NB*SP + 255)/256, 256, 0, stream>>>(x, eps, Wmix, ov,
                                                    (float*)d_out);
}
